// LuongAttention_60679297958493
// MI455X (gfx1250) — matrix operand order repaired and checked by reading, not compiled, not run on hardware
//
#include <hip/hip_runtime.h>

typedef __attribute__((ext_vector_type(16))) __bf16 v16bf;
typedef __attribute__((ext_vector_type(8)))  float  v8f;
typedef unsigned int  uint;
typedef unsigned short ushort;

#define DD 512
#define SS 2048
#define BB 64

// ---------------------------------------------------------------------------
// Kernel 1: dec_proj[b,n] = sum_e ht[b,e] * W_dec[n,e] + W_b[n]
// ---------------------------------------------------------------------------
__global__ __launch_bounds__(128) void dp_kernel(const float* __restrict__ ht,
                                                 const float* __restrict__ Ww,
                                                 const float* __restrict__ Wb,
                                                 float* __restrict__ dp)
{
    const int id = blockIdx.x * 128 + threadIdx.x;   // 0 .. 64*512-1
    const int b  = id >> 9;
    const int n  = id & 511;
    const float* __restrict__ h = ht + b * DD;
    const float* __restrict__ w = Ww + (size_t)n * (2 * DD) + DD;  // W_dec row n
    float s = 0.f;
    for (int e = 0; e < DD; e += 4) {
        float4 hv = *(const float4*)(h + e);
        float4 wv = *(const float4*)(w + e);
        s += hv.x * wv.x + hv.y * wv.y + hv.z * wv.z + hv.w * wv.w;
    }
    dp[id] = s + Wb[n];
}

// ---------------------------------------------------------------------------
// Kernel 2: convert W_enc (fp32, row n, col k) into bf16 B-fragments laid out
// in the contiguous k order (k = 16*half + s) that score_kernel's A fragments also use:
//   fragment (j = n-chunk 0..31, t = k-step 0..15):
//     lane L: n = j*16 + (L&15), k-base = t*32 + (L>>4)*16,
//     16 consecutive bf16 K-values packed into 8 dwords (32 bytes).
//   stored contiguously: offset = ((j*16 + t)*32 + L) * 32 bytes
// ---------------------------------------------------------------------------
__global__ __launch_bounds__(128) void wswz_kernel(const float* __restrict__ Ww,
                                                   uint* __restrict__ wswz)
{
    const int id   = blockIdx.x * 128 + threadIdx.x;  // 0 .. 32*16*32-1
    const int lane = id & 31;
    const int t    = (id >> 5) & 15;
    const int j    = id >> 9;
    const int n    = j * 16 + (lane & 15);
    const int kb   = t * 32 + (lane >> 4) * 16;
    const float* __restrict__ src = Ww + (size_t)n * (2 * DD) + kb;  // W_enc[n, kb..kb+15]
    uint outq[8];
#pragma unroll
    for (int i = 0; i < 8; ++i) {
        __bf16 lo = (__bf16)src[2 * i];
        __bf16 hi = (__bf16)src[2 * i + 1];
        const ushort ulo = __builtin_bit_cast(ushort, lo);
        const ushort uhi = __builtin_bit_cast(ushort, hi);
        outq[i] = (uint)ulo | ((uint)uhi << 16);
    }
    uint* dst = wswz + (size_t)id * 8;
    *(uint4*)(dst)     = *(const uint4*)&outq[0];
    *(uint4*)(dst + 4) = *(const uint4*)&outq[4];
}

// ---------------------------------------------------------------------------
// Kernel 3 (main): fused  scores = (tanh(enc @ W_enc^T + dp) @ V) + V_b
// WG = 256 threads (8 waves), M-tile = 128 rows (wave w owns rows 16w..16w+15).
// A-operand: wave's 16x512 enc rows, fp32->bf16 once, 16 WMMA A-fragments kept
// in 128 VGPRs (enc read from HBM exactly once in this pass).
// B-operand: pre-swizzled bf16 W fragments, DMA'd into a double-buffered LDS
// tile with GLOBAL_LOAD_ASYNC_TO_LDS_B128 (ASYNCcnt-tracked), prefetching
// chunk j+1 while chunk j's 16 v_wmma_f32_16x16x32_bf16 execute.
// ---------------------------------------------------------------------------
union FragB { uint4 q[2]; v16bf v; };

__global__ __launch_bounds__(256) void score_kernel(const float* __restrict__ enc,
                                                    const float* __restrict__ dp,
                                                    const float* __restrict__ Vw,
                                                    const float* __restrict__ Vb,
                                                    const uint4* __restrict__ wswz,
                                                    float* __restrict__ scores)
{
    __shared__ uint4 sB[2048];    // double buffer: 2 x (16 t * 32 lanes * 32B) = 32 KB
    __shared__ float sDP[DD];
    __shared__ float sV[DD];

    const int tid  = threadIdx.x;
    const int wave = tid >> 5;
    const int L    = tid & 31;
    const int half = L >> 4;      // which 16-lane half
    const int col  = L & 15;

    const size_t m0 = (size_t)blockIdx.x * 128;   // first global row of tile
    const int b     = (int)(m0 >> 11);            // all 128 rows share one batch (2048%128==0)

    sDP[tid]       = dp[b * DD + tid];
    sDP[tid + 256] = dp[b * DD + tid + 256];
    sV[tid]        = Vw[tid];
    sV[tid + 256]  = Vw[tid + 256];

    const uint sb_base = (uint)(uintptr_t)&sB[0];

    // async DMA: copy 64B/thread of chunk JJ into LDS buffer NB (4 x b128;
    // INST_OFFSET applies to both the LDS and global addresses)
#define STAGE_ASYNC(JJ, NB)                                                          \
    do {                                                                             \
        const unsigned long long ga_ = (unsigned long long)(uintptr_t)               \
            ((const char*)wswz + ((size_t)(JJ) << 14) + ((size_t)tid << 6));         \
        const uint la_ = sb_base + ((uint)(NB) << 14) + ((uint)tid << 6);            \
        asm volatile("global_load_async_to_lds_b128 %0, %1, off\n"                   \
                     "global_load_async_to_lds_b128 %0, %1, off offset:16\n"         \
                     "global_load_async_to_lds_b128 %0, %1, off offset:32\n"         \
                     "global_load_async_to_lds_b128 %0, %1, off offset:48\n"         \
                     :: "v"(la_), "v"(ga_) : "memory");                              \
    } while (0)

    // ---- load & convert this wave's A tile (16 rows x 512 K) into registers
    const int mrow = (int)m0 + wave * 16 + col;
    const float* __restrict__ arow = enc + (size_t)mrow * DD;

    v16bf afrag[16];
#pragma unroll
    for (int t = 0; t < 16; ++t) {
        const int k0 = t * 32 + half * 16;  // lanes<16: K 0..15 ; lanes>=16: 16..31
        float4 f0 = *(const float4*)(arow + k0);
        float4 f1 = *(const float4*)(arow + k0 + 4);
        float4 f2 = *(const float4*)(arow + k0 + 8);
        float4 f3 = *(const float4*)(arow + k0 + 12);
        v16bf a;
        a[0]  = (__bf16)f0.x; a[1]  = (__bf16)f0.y; a[2]  = (__bf16)f0.z; a[3]  = (__bf16)f0.w;
        a[4]  = (__bf16)f1.x; a[5]  = (__bf16)f1.y; a[6]  = (__bf16)f1.z; a[7]  = (__bf16)f1.w;
        a[8]  = (__bf16)f2.x; a[9]  = (__bf16)f2.y; a[10] = (__bf16)f2.z; a[11] = (__bf16)f2.w;
        a[12] = (__bf16)f3.x; a[13] = (__bf16)f3.y; a[14] = (__bf16)f3.z; a[15] = (__bf16)f3.w;
        afrag[t] = a;
    }

    float ps[8];
#pragma unroll
    for (int r = 0; r < 8; ++r) ps[r] = 0.f;

    // prime the pipeline: chunk 0 -> buffer 0
    STAGE_ASYNC(0, 0);

    // ---- loop over the 32 column chunks of W (N = 512), double-buffered
    for (int j = 0; j < 32; ++j) {
        const int buf  = j & 1;
        const int nbuf = buf ^ 1;

        __syncthreads();                     // chunk j-1 fully consumed: nbuf reusable
        STAGE_ASYNC((j + 1) & 31, nbuf);     // prefetch next chunk (j=31 wraps; harmless)
        asm volatile("s_wait_asynccnt 0x4" ::: "memory");  // oldest 4 (chunk j) landed
        __syncthreads();                     // every wave's slice of chunk j visible

        v8f acc = {};
        const int bbase = buf * 1024 + L * 2;
#pragma unroll
        for (int t = 0; t < 16; ++t) {
            FragB fb;
            fb.q[0] = sB[bbase + t * 64];
            fb.q[1] = sB[bbase + t * 64 + 1];
            acc = __builtin_amdgcn_wmma_f32_16x16x32_bf16(
                      /*neg_a=*/false, afrag[t],
                      /*neg_b=*/false, fb.v,
                      /*c_mod=*/(short)0, acc,
                      /*reuse_a=*/false, /*reuse_b=*/false);
        }

        // epilogue for this 16x16 tile: add dec-proj, tanh, dot with V
        const int n    = j * 16 + col;          // C/D layout: column = lane&15
        const float dv = sDP[n];
        const float vv = sV[n];
#pragma unroll
        for (int r = 0; r < 8; ++r)             // row = r + 8*(lane>>4)
            ps[r] += tanhf(acc[r] + dv) * vv;
    }
#undef STAGE_ASYNC

    // drain the wrapped redundant prefetch before the wave exits the LDS scope
    asm volatile("s_wait_asynccnt 0x0" ::: "memory");

    // ---- reduce score partials across the 16 lanes of each half, write out
    const float vb = Vb[0];
#pragma unroll
    for (int r = 0; r < 8; ++r) {
        float v = ps[r];
        v += __shfl_xor(v, 1);
        v += __shfl_xor(v, 2);
        v += __shfl_xor(v, 4);
        v += __shfl_xor(v, 8);
        if (col == 0) {
            const int m = (int)m0 + wave * 16 + r + half * 8;
            scores[m] = v + vb;
        }
    }
}

// ---------------------------------------------------------------------------
// Kernel 4: softmax over S per batch + weighted sum c_t = softmax(scores) @ enc
// grid (64, 4): each block handles one batch b and 128 d-columns (fp32,
// coalesced; enc's second and final HBM pass).
// ---------------------------------------------------------------------------
__global__ __launch_bounds__(128) void ctx_kernel(const float* __restrict__ enc,
                                                  const float* __restrict__ scores,
                                                  float* __restrict__ out)
{
    __shared__ float red[128];
    __shared__ float wbuf[SS];
    const int b = blockIdx.x;
    const int t = threadIdx.x;
    const int d = blockIdx.y * 128 + t;
    const float* __restrict__ sc = scores + (size_t)b * SS;

    float mx = -3.402823466e38f;
    for (int i = t; i < SS; i += 128) mx = fmaxf(mx, sc[i]);
    red[t] = mx; __syncthreads();
    for (int o = 64; o > 0; o >>= 1) { if (t < o) red[t] = fmaxf(red[t], red[t + o]); __syncthreads(); }
    mx = red[0]; __syncthreads();

    float sm = 0.f;
    for (int i = t; i < SS; i += 128) { const float w = expf(sc[i] - mx); wbuf[i] = w; sm += w; }
    red[t] = sm; __syncthreads();
    for (int o = 64; o > 0; o >>= 1) { if (t < o) red[t] += red[t + o]; __syncthreads(); }
    const float inv = 1.0f / red[0];

    const float* __restrict__ eb = enc + (size_t)b * SS * DD + d;
    float a0 = 0.f, a1 = 0.f, a2 = 0.f, a3 = 0.f;
    for (int s = 0; s < SS; s += 4) {
        a0 += wbuf[s]     * eb[(size_t)(s)     * DD];
        a1 += wbuf[s + 1] * eb[(size_t)(s + 1) * DD];
        a2 += wbuf[s + 2] * eb[(size_t)(s + 2) * DD];
        a3 += wbuf[s + 3] * eb[(size_t)(s + 3) * DD];
    }
    out[b * DD + d] = (a0 + a1 + a2 + a3) * inv;
}

// ---------------------------------------------------------------------------
// launch:  inputs (all fp32): enc_outs, ht, W_w, W_b, V_w, V_b
// workspace: [0, 512KB) scores | [512KB, 640KB) dec_proj | [640KB, 1152KB) W bf16 swizzled
// ---------------------------------------------------------------------------
extern "C" void kernel_launch(void* const* d_in, const int* in_sizes, int n_in,
                              void* d_out, int out_size, void* d_ws, size_t ws_size,
                              hipStream_t stream)
{
    (void)in_sizes; (void)n_in; (void)out_size; (void)ws_size;
    const float* enc = (const float*)d_in[0];
    const float* ht  = (const float*)d_in[1];
    const float* Ww  = (const float*)d_in[2];
    const float* Wb  = (const float*)d_in[3];
    const float* Vw  = (const float*)d_in[4];
    const float* Vb  = (const float*)d_in[5];
    float* out = (float*)d_out;

    float* ws_scores = (float*)d_ws;                              // 64*2048 fp32
    float* ws_dp     = (float*)((char*)d_ws + (512u << 10));      // 64*512  fp32
    uint*  ws_w      = (uint*)((char*)d_ws + (640u << 10));       // 512*512 bf16 swizzled

    dp_kernel<<<(BB * DD) / 128, 128, 0, stream>>>(ht, Ww, Wb, ws_dp);
    wswz_kernel<<<(32 * 16 * 32) / 128, 128, 0, stream>>>(Ww, ws_w);
    score_kernel<<<(BB * SS) / 128, 256, 0, stream>>>(enc, ws_dp, Vw, Vb,
                                                      (const uint4*)ws_w, ws_scores);
    dim3 g(BB, DD / 128);
    ctx_kernel<<<g, 128, 0, stream>>>(enc, ws_scores, out);
}
